// MBDSCA_59734405153188
// MI455X (gfx1250) — matrix operand order repaired and checked by reading, not compiled, not run on hardware
//
#include <hip/hip_runtime.h>
#include <hip/hip_bf16.h>
#include <stdint.h>

typedef _Float16 f16;
typedef __attribute__((ext_vector_type(16))) _Float16 v16h;
typedef __attribute__((ext_vector_type(8)))  _Float16 v8h;
typedef __attribute__((ext_vector_type(4)))  _Float16 v4h;
typedef __attribute__((ext_vector_type(8)))  float    v8f;

#define BATCH 512
#define TT    16
#define NNODE 22
#define CC    64
#define DD    1024          // TT*CC
#define NPAD  32

#define XH_STRIDE  1040     // halves per row of xh  (2080 B, 16B aligned)
#define XHT_STRIDE 40       // halves per row of xhT (80 B, 16B aligned)
#define ATT_STRIDE 40       // halves per row of attf

// byte offsets into dynamic LDS
#define OFF_XH    0                       // 32*1040*2  = 66560
#define OFF_XHT   66560                   // 1024*40*2  = 81920
#define OFF_SIM   148480                  // 32*32*4    = 4096
#define OFF_ATT   152576                  // 32*40*2    = 2560
#define OFF_AF    155136                  // 32*32*4    = 4096
#define OFF_INVN  159232                  // 32*4 = 128
#define OFF_INVD  159360                  // 32*4 = 128
#define SMEM_BYTES 159488

__device__ __forceinline__ v16h cat8(v8h lo, v8h hi) {
  return __builtin_shufflevector(lo, hi, 0,1,2,3,4,5,6,7,8,9,10,11,12,13,14,15);
}
__device__ __forceinline__ v8f wmma16(v16h a, v16h b, v8f c) {
  // v_wmma_f32_16x16x32_f16 : D = A(16x32 f16) * B(32x16 f16) + C(f32)
  return __builtin_amdgcn_wmma_f32_16x16x32_f16(false, a, false, b, (short)0, c,
                                                false, false);
}

__global__ __launch_bounds__(256)
void MBDSCA_gcn_kernel(const float* __restrict__ x,
                       const float* __restrict__ adj,
                       const float* __restrict__ temps,
                       const float* __restrict__ flog,
                       float* __restrict__ out) {
  extern __shared__ __align__(16) char smem[];
  f16*   xh   = (f16*)(smem + OFF_XH);    // [32][XH_STRIDE]  f16 current features
  f16*   xhT  = (f16*)(smem + OFF_XHT);   // [1024][XHT_STRIDE] transposed copy
  float* sim  = (float*)(smem + OFF_SIM); // [32][32] masked Gram
  f16*   attf = (f16*)(smem + OFF_ATT);   // [32][ATT_STRIDE] fused attention
  float* Afull= (float*)(smem + OFF_AF);  // [32][32] normalized adjacency (0-padded)
  float* invn = (float*)(smem + OFF_INVN);// [32] 1/(||row||+1e-6), 0 for pads
  float* invd = (float*)(smem + OFF_INVD);// [32] degree^-1/2

  const int tid  = threadIdx.x;
  const int b    = blockIdx.x;
  const int wave = tid >> 5;
  const int lane = tid & 31;
  const int lr   = lane & 15;
  const int sel  = lane >> 4;
  const float* xg = x + (size_t)b * (TT * NNODE * CC);

  // ---------------- phase 0: normalized adjacency (every block; tiny) -------
  for (int i = tid; i < NPAD * NPAD; i += 256) Afull[i] = 0.0f;
  __syncthreads();
  for (int idx = tid; idx < NNODE * NNODE; idx += 256) {
    int n = idx / NNODE, m = idx % NNODE;
    float v = adj[idx];
    v = fminf(fmaxf(v, 0.0f), 1.0f);          // clip(adj,0,1)
    if (n == m) v += 1.0f;                    // + I
    v = fmaxf(v, 1e-8f);                      // clip(.,1e-8,inf)
    Afull[n * NPAD + m] = v;
  }
  __syncthreads();
  if (tid < NNODE) {
    float s = 0.0f;
    for (int m = 0; m < NNODE; ++m) s += Afull[tid * NPAD + m];
    s = fmaxf(s, 1e-8f);
    invd[tid] = fminf(rsqrtf(s), 100.0f);     // clip(rs^-0.5, 0, 100)
  }
  __syncthreads();
  for (int idx = tid; idx < NNODE * NNODE; idx += 256) {
    int n = idx / NNODE, m = idx % NNODE;
    Afull[n * NPAD + m] *= invd[n] * invd[m];
  }
  __syncthreads();
  if (b == 0) {  // second output: A_norm [22][22]
    for (int idx = tid; idx < NNODE * NNODE; idx += 256)
      out[(size_t)BATCH * NNODE * DD + idx] =
          Afull[(idx / NNODE) * NPAD + (idx % NNODE)];
  }

  // ---------------- phase 1: load + transpose x -> xh (f16) & xhT ----------
  for (int i = tid; i < OFF_SIM / 4; i += 256)    // zero xh + xhT (pads stay 0)
    ((uint32_t*)smem)[i] = 0u;
  __syncthreads();
  for (int idx = tid; idx < (TT * NNODE * CC) / 4; idx += 256) {
    int f = idx * 4;
    int c = f & 63;
    int nm = f >> 6;
    int n = nm % NNODE;
    int t = nm / NNODE;
    const float4 v = *(const float4*)(xg + f);
    int d = t * CC + c;
    v4h h; h.x = (f16)v.x; h.y = (f16)v.y; h.z = (f16)v.z; h.w = (f16)v.w;
    *(v4h*)(xh + n * XH_STRIDE + d) = h;
    xhT[(d + 0) * XHT_STRIDE + n] = h.x;
    xhT[(d + 1) * XHT_STRIDE + n] = h.y;
    xhT[(d + 2) * XHT_STRIDE + n] = h.z;
    xhT[(d + 3) * XHT_STRIDE + n] = h.w;
  }
  __syncthreads();

  // ---------------- layers --------------------------------------------------
  for (int layer = 0; layer < 2; ++layer) {
    // zero sim / attf
    for (int i = tid; i < NPAD * NPAD; i += 256) sim[i] = 0.0f;
    for (int i = tid; i < NPAD * ATT_STRIDE; i += 256) attf[i] = (f16)0.0f;
    __syncthreads();

    // --- row inverse norms (f16-rounded features; unit-scale, OK) ----------
    for (int r = 0; r < 4; ++r) {
      int n = wave + 8 * r;
      if (n < NPAD) {
        float s = 0.0f;
        if (n < NNODE) {
          for (int dd = lane; dd < DD; dd += 32) {
            float xv = (float)xh[n * XH_STRIDE + dd];
            s += xv * xv;
          }
          for (int off = 16; off; off >>= 1) s += __shfl_xor(s, off, 32);
        }
        if (lane == 0) invn[n] = (n < NNODE) ? 1.0f / (sqrtf(s) + 1e-6f) : 0.0f;
      }
    }
    __syncthreads();

    // --- Gram GEMM: G = Xh * Xh^T (32x32 padded), then mask/normalize ------
    {
      const int tile = wave & 3;         // 4 output tiles (2x2)
      const int kh   = wave >> 2;        // 2 waves split K=1024
      const int ti   = tile >> 1, tj = tile & 1;
      const int rowA = ti * 16 + lr;
      const int rowB = tj * 16 + lr;
      v8f acc = {};
      for (int kk = 0; kk < 16; ++kk) {
        int k0 = kh * 512 + kk * 32;
        // A fragment: row rowA, K halves {k0+sel*8 .. +7, k0+16+sel*8 .. +7}
        v8h a0 = *(const v8h*)(xh + rowA * XH_STRIDE + k0 + sel * 8);
        v8h a1 = *(const v8h*)(xh + rowA * XH_STRIDE + k0 + 16 + sel * 8);
        // B fragment (B = Xh^T): column rowB = row rowB of Xh, K halves {k0+sel*8 .. +7, k0+16+sel*8 .. +7}
        v8h b0 = *(const v8h*)(xh + rowB * XH_STRIDE + k0 + sel * 8);
        v8h b1 = *(const v8h*)(xh + rowB * XH_STRIDE + k0 + 16 + sel * 8);
        acc = wmma16(cat8(a0, a1), cat8(b0, b1), acc);
      }
#pragma unroll
      for (int r = 0; r < 8; ++r) {
        int gn = ti * 16 + r + 8 * sel;   // C/D layout: lanes>=16 hold M=r+8
        int gm = tj * 16 + lr;
        float v = acc[r] * invn[gn] * invn[gm] * Afull[gn * NPAD + gm];
        atomicAdd(&sim[gn * NPAD + gm], v);   // ds_add_f32, 2 K-halves combine
      }
    }
    __syncthreads();

    // --- branches: temp scale + diag boost + topK + softmax, alpha-fused ---
    if (tid < NNODE) {
      const int n = tid;
      const int K = (layer == 0) ? 10 : 3;   // annealed top-k
      float l0 = flog[layer * 3 + 0], l1 = flog[layer * 3 + 1],
            l2 = flog[layer * 3 + 2];
      float mx = fmaxf(l0, fmaxf(l1, l2));
      float e0 = __expf(l0 - mx), e1 = __expf(l1 - mx), e2 = __expf(l2 - mx);
      float ai = 1.0f / (e0 + e1 + e2);
      float alpha[3] = {e0 * ai, e1 * ai, e2 * ai};

      float rowv[NNODE], fused[NNODE];
      for (int m = 0; m < NNODE; ++m) { rowv[m] = sim[n * NPAD + m]; fused[m] = 0.0f; }
      for (int k = 0; k < 3; ++k) {
        float tk = fminf(fmaxf(temps[k], 0.1f), 10.0f);
        float a[NNODE];
        for (int m = 0; m < NNODE; ++m)
          a[m] = rowv[m] / tk + ((m == n) ? 0.1f : 0.0f);
        uint32_t chosen = 0;
        for (int kk = 0; kk < K; ++kk) {       // selection top-K (first-index ties)
          float best = -1e30f; int bi = 0;
          for (int m = 0; m < NNODE; ++m)
            if (!((chosen >> m) & 1u) && a[m] > best) { best = a[m]; bi = m; }
          chosen |= 1u << bi;
        }
        float vmax = -1e30f;
        for (int m = 0; m < NNODE; ++m) {
          float v = ((chosen >> m) & 1u) ? a[m] : 0.0f;  // zeros still in softmax
          vmax = fmaxf(vmax, v);
        }
        float p[NNODE], ssum = 0.0f;
        for (int m = 0; m < NNODE; ++m) {
          float v = ((chosen >> m) & 1u) ? a[m] : 0.0f;
          p[m] = __expf(v - vmax);
          ssum += p[m];
        }
        float inv = alpha[k] / ssum;
        for (int m = 0; m < NNODE; ++m) fused[m] += p[m] * inv;
      }
      for (int m = 0; m < NNODE; ++m)
        attf[n * ATT_STRIDE + m] = (f16)fused[m];  // cols 22..31 stay zero
    }
    __syncthreads();

    // --- aggregation GEMM: Xnew = AttFused(32x32) * Xh(32x1024) ------------
    {
      // A fragments constant over the layer: rows 0-15 and 16-31 of attf
      v8h a00 = *(const v8h*)(attf + (0 * 16 + lr) * ATT_STRIDE + sel * 8);
      v8h a01 = *(const v8h*)(attf + (0 * 16 + lr) * ATT_STRIDE + 16 + sel * 8);
      v8h a10 = *(const v8h*)(attf + (1 * 16 + lr) * ATT_STRIDE + sel * 8);
      v8h a11 = *(const v8h*)(attf + (1 * 16 + lr) * ATT_STRIDE + 16 + sel * 8);
      v16h afr0 = cat8(a00, a01);
      v16h afr1 = cat8(a10, a11);

      for (int dt = 0; dt < 8; ++dt) {
        int dtile = wave * 8 + dt;                 // 64 feature tiles / 8 waves
        int drow  = dtile * 16 + lr;               // B col = feature index
        v8h b0 = *(const v8h*)(xhT + drow * XHT_STRIDE + sel * 8);
        v8h b1 = *(const v8h*)(xhT + drow * XHT_STRIDE + 16 + sel * 8);
        v16h bfr = cat8(b0, b1);
        v8f z0 = {}, z1 = {};
        v8f acc0 = wmma16(afr0, bfr, z0);          // node rows 0-15
        v8f acc1 = wmma16(afr1, bfr, z1);          // node rows 16-31
        // epilogue AFTER both rtiles read xhT rows of this dtile (no WAR)
#pragma unroll
        for (int r = 0; r < 8; ++r) {
          int dcol = dtile * 16 + lr;
          int n0 = r + 8 * sel;
          int n1 = 16 + r + 8 * sel;
          if (layer == 0) {
            float v0 = fmaxf(acc0[r], 0.0f);       // relu (layer < last)
            f16 h0 = (f16)v0;
            xh[n0 * XH_STRIDE + dcol] = h0;
            xhT[dcol * XHT_STRIDE + n0] = h0;
            if (n1 < NNODE) {
              float v1 = fmaxf(acc1[r], 0.0f);
              f16 h1 = (f16)v1;
              xh[n1 * XH_STRIDE + dcol] = h1;
              xhT[dcol * XHT_STRIDE + n1] = h1;
            }
          } else {                                 // final: + x0 residual, store
            int t = dcol >> 6, c = dcol & 63;
            float x00 = xg[(t * NNODE + n0) * CC + c];
            out[((size_t)b * NNODE + n0) * DD + dcol] = acc0[r] + x00;
            if (n1 < NNODE) {
              float x01 = xg[(t * NNODE + n1) * CC + c];
              out[((size_t)b * NNODE + n1) * DD + dcol] = acc1[r] + x01;
            }
          }
        }
      }
    }
    __syncthreads();   // attf/xh/xhT settled before next layer
  }
}

extern "C" void kernel_launch(void* const* d_in, const int* in_sizes, int n_in,
                              void* d_out, int out_size, void* d_ws, size_t ws_size,
                              hipStream_t stream) {
  (void)in_sizes; (void)n_in; (void)out_size; (void)d_ws; (void)ws_size;
  const float* x     = (const float*)d_in[0];
  const float* adj   = (const float*)d_in[1];
  const float* temps = (const float*)d_in[2];
  const float* flog  = (const float*)d_in[3];
  float* out = (float*)d_out;
  MBDSCA_gcn_kernel<<<dim3(BATCH), dim3(256), SMEM_BYTES, stream>>>(
      x, adj, temps, flog, out);
}
